// Attention_66030827208775
// MI455X (gfx1250) — matrix operand order repaired and checked by reading, not compiled, not run on hardware
//
#include <hip/hip_runtime.h>

typedef __attribute__((ext_vector_type(16))) _Float16 v16h;
typedef __attribute__((ext_vector_type(8)))  _Float16 v8h;
typedef __attribute__((ext_vector_type(8)))  float    v8f;

constexpr int kS = 1024, kT = 1024, kB = 32, kD = 1024, kA = 1024;
constexpr int BM = 128, BN = 128, BK = 32;      // block tile, K-chunk
constexpr int NIT = kD / BK;                    // 32 K-steps

// ---- WMMA fragment loaders (CDNA5 ISA 7.12.2, 16-bit, wave32) ----
// A 16x32: lane=16h+r. VGPR0-3: K=h*8+{0..7}; VGPR4-7: K=16+h*8+{0..7}.
__device__ inline v16h load_frag_a(const _Float16* base, int lane) {
    const int r = lane & 15, h = lane >> 4;
    const unsigned int* p = (const unsigned int*)(base + r * BK);
    union { v16h v; unsigned int u[8]; } f;
#pragma unroll
    for (int j = 0; j < 4; ++j) {
        f.u[j]     = p[h * 4 + j];
        f.u[j + 4] = p[8 + h * 4 + j];
    }
    return f.v;
}
// B 32x16: lane=16h+c. VGPR0-3: K=h*8+{0..7}; VGPR4-7: K=16+h*8+{0..7}. LDS tile stored [n][k].
__device__ inline v16h load_frag_b(const _Float16* base, int lane) {
    const int c = lane & 15, h = lane >> 4;
    const unsigned int* p = (const unsigned int*)(base + c * BK);
    union { v16h v; unsigned int u[8]; } f;
#pragma unroll
    for (int j = 0; j < 4; ++j) {
        f.u[j]     = p[h * 4 + j];
        f.u[j + 4] = p[8 + h * 4 + j];
    }
    return f.v;
}

__device__ inline v8f wmma_step(v16h a, v16h b, v8f c) {
    return __builtin_amdgcn_wmma_f32_16x16x32_f16(false, a, false, b, (short)0, c,
                                                  false, false);
}

// 8 WMMAs: wave tile 32(M) x 64(N) out of a 128x128 block tile (8 waves 4x2)
__device__ inline void mma_tile(const _Float16* As, const _Float16* Bs,
                                int wm, int wn, int lane, v8f c[2][4]) {
    v16h a0 = load_frag_a(As + (wm * 32) * BK,      lane);
    v16h a1 = load_frag_a(As + (wm * 32 + 16) * BK, lane);
    v16h bfr[4];
#pragma unroll
    for (int j = 0; j < 4; ++j)
        bfr[j] = load_frag_b(Bs + (wn * 64 + j * 16) * BK, lane);
#pragma unroll
    for (int j = 0; j < 4; ++j) {
        c[0][j] = wmma_step(a0, bfr[j], c[0][j]);
        c[1][j] = wmma_step(a1, bfr[j], c[1][j]);
    }
}

// ---- async global->LDS staging of a 128x32 f16 tile (k-contiguous rows) ----
// 512 x 16B chunks; 2 global_load_async_to_lds_b128 per thread (ASYNCcnt +=4
// per wave per A+B pair).
__device__ inline void stage128x32(unsigned int lds_base, const _Float16* g,
                                   size_t ld, int tid) {
#pragma unroll
    for (int i = 0; i < 2; ++i) {
        int e = tid + 256 * i;
        int c4 = e & 3, r = e >> 2;
        unsigned int lp = lds_base + (unsigned int)(r * (BK * 2) + c4 * 16);
        const void* gp = (const void*)(g + (size_t)r * ld + c4 * 8);
        asm volatile("global_load_async_to_lds_b128 %0, %1, off"
                     :: "v"(lp), "v"(gp) : "memory");
    }
}
__device__ inline void wait_async(int n) {
    if (n) asm volatile("s_wait_asynccnt 0x4" ::: "memory");
    else   asm volatile("s_wait_asynccnt 0x0" ::: "memory");
}

// ============================================================================
// Kernel 0: f32 -> f16 bulk convert (8 elements / thread, b128 in, b128 out)
// ============================================================================
__global__ __launch_bounds__(256) void cvt_f16(const float* __restrict__ in,
                                               _Float16* __restrict__ out) {
    size_t i = ((size_t)blockIdx.x * 256 + threadIdx.x) * 8;
    const float4* p = (const float4*)(in + i);
    float4 x = p[0], y = p[1];
    v8h o;
    o[0] = (_Float16)x.x; o[1] = (_Float16)x.y; o[2] = (_Float16)x.z; o[3] = (_Float16)x.w;
    o[4] = (_Float16)y.x; o[5] = (_Float16)y.y; o[6] = (_Float16)y.z; o[7] = (_Float16)y.w;
    *(v8h*)(out + i) = o;
}

// ============================================================================
// Kernel 1: projection  Y[b][m][n] = sum_k X[m][b][k] * W[n][k]   (all f16 in)
//   X: (rows,kB,kD) f16 ; W: (kA,kD) f16 ; Y: (kB,rows,kA) f16
// Double-buffered async staging.
// ============================================================================
__global__ __launch_bounds__(256) void proj_f16(const _Float16* __restrict__ X,
                                                const _Float16* __restrict__ W,
                                                _Float16* __restrict__ Y) {
    __shared__ __align__(16) _Float16 As[2][BM][BK];
    __shared__ __align__(16) _Float16 Bs[2][BN][BK];
    const int b = blockIdx.z, m0 = blockIdx.y * BM, n0 = blockIdx.x * BN;
    const int tid = threadIdx.x, lane = tid & 31, wave = tid >> 5;
    const int wm = wave >> 1, wn = wave & 1;
    const unsigned int asb = (unsigned int)(uintptr_t)&As[0][0][0];
    const unsigned int bsb = (unsigned int)(uintptr_t)&Bs[0][0][0];
    const _Float16* Ag = X + (size_t)m0 * (kB * kD) + (size_t)b * kD;  // ld = kB*kD
    const _Float16* Bg = W + (size_t)n0 * kD;                          // ld = kD
    v8f c[2][4];
#pragma unroll
    for (int a = 0; a < 2; ++a)
#pragma unroll
        for (int j = 0; j < 4; ++j) c[a][j] = (v8f){};

    stage128x32(asb, Ag, kB * kD, tid);
    stage128x32(bsb, Bg, kD, tid);
    for (int it = 0; it < NIT; ++it) {
        const int cur = it & 1;
        if (it + 1 < NIT) {
            const int nxt = cur ^ 1, kc = (it + 1) * BK;
            stage128x32(asb + nxt * (BM * BK * 2), Ag + kc, kB * kD, tid);
            stage128x32(bsb + nxt * (BN * BK * 2), Bg + kc, kD, tid);
            wait_async(1);          // retire current buffer's 4 async groups
        } else {
            wait_async(0);
        }
        __syncthreads();
        mma_tile(&As[cur][0][0], &Bs[cur][0][0], wm, wn, lane, c);
        __syncthreads();
    }
    const int h = lane >> 4, nl = lane & 15;
    _Float16* Yb = Y + (size_t)b * kS * kA;
#pragma unroll
    for (int ai = 0; ai < 2; ++ai)
#pragma unroll
        for (int j = 0; j < 4; ++j)
#pragma unroll
            for (int r = 0; r < 8; ++r) {
                int row = m0 + wm * 32 + ai * 16 + r + 8 * h;
                int col = n0 + wn * 64 + j * 16 + nl;
                Yb[(size_t)row * kA + col] = (_Float16)c[ai][j][r];
            }
}

// ============================================================================
// Kernel 2: scores[s][t][b] (f32) = sum_a ts[b][s][a] * tt[b][t][a]
// ============================================================================
__global__ __launch_bounds__(256) void scores_kernel(const _Float16* __restrict__ TS,
                                                     const _Float16* __restrict__ TT,
                                                     float* __restrict__ SC) {
    __shared__ __align__(16) _Float16 As[2][BM][BK];
    __shared__ __align__(16) _Float16 Bs[2][BN][BK];
    const int b = blockIdx.z, m0 = blockIdx.y * BM, n0 = blockIdx.x * BN;
    const int tid = threadIdx.x, lane = tid & 31, wave = tid >> 5;
    const int wm = wave >> 1, wn = wave & 1;
    const unsigned int asb = (unsigned int)(uintptr_t)&As[0][0][0];
    const unsigned int bsb = (unsigned int)(uintptr_t)&Bs[0][0][0];
    const _Float16* Ag = TS + (size_t)b * kS * kA + (size_t)m0 * kA;
    const _Float16* Bg = TT + (size_t)b * kT * kA + (size_t)n0 * kA;
    v8f c[2][4];
#pragma unroll
    for (int a = 0; a < 2; ++a)
#pragma unroll
        for (int j = 0; j < 4; ++j) c[a][j] = (v8f){};

    stage128x32(asb, Ag, kA, tid);
    stage128x32(bsb, Bg, kA, tid);
    for (int it = 0; it < NIT; ++it) {
        const int cur = it & 1;
        if (it + 1 < NIT) {
            const int nxt = cur ^ 1, kc = (it + 1) * BK;
            stage128x32(asb + nxt * (BM * BK * 2), Ag + kc, kA, tid);
            stage128x32(bsb + nxt * (BN * BK * 2), Bg + kc, kA, tid);
            wait_async(1);
        } else {
            wait_async(0);
        }
        __syncthreads();
        mma_tile(&As[cur][0][0], &Bs[cur][0][0], wm, wn, lane, c);
        __syncthreads();
    }
    const int h = lane >> 4, nl = lane & 15;
#pragma unroll
    for (int ai = 0; ai < 2; ++ai)
#pragma unroll
        for (int j = 0; j < 4; ++j)
#pragma unroll
            for (int r = 0; r < 8; ++r) {
                int s = m0 + wm * 32 + ai * 16 + r + 8 * h;
                int t = n0 + wn * 64 + j * 16 + nl;
                SC[(size_t)s * (kT * kB) + (size_t)t * kB + b] = c[ai][j][r];
            }
}

// ============================================================================
// Kernel 3: softmax over S axis of scores (S,T,B), in place (coalesced).
// ============================================================================
__global__ __launch_bounds__(256) void softmax_s(float* __restrict__ SC) {
    const int col = blockIdx.x * 256 + threadIdx.x;   // flat (t,b)
    const int stride = kT * kB;
    float m = -3.402823466e38f, sum = 0.0f;
    for (int s = 0; s < kS; ++s) {
        float x  = SC[(size_t)s * stride + col];
        float nm = fmaxf(m, x);
        sum = sum * __expf(m - nm) + __expf(x - nm);
        m = nm;
    }
    const float inv = 1.0f / sum;
    for (int s = 0; s < kS; ++s) {
        size_t idx = (size_t)s * stride + col;
        SC[idx] = __expf(SC[idx] - m) * inv;
    }
}

// ============================================================================
// Kernel 4: out[t][b][d] (f32) = sum_s weights[s][t][b] * srcH[s][b][d]
// A operand is k-gather-strided -> scalar staging; B reads f16 source copy.
// ============================================================================
__global__ __launch_bounds__(256) void out_kernel(const float* __restrict__ Wgt,      // (S,T,B) f32
                                                  const _Float16* __restrict__ SrcH,  // (S,B,D) f16
                                                  float* __restrict__ Out) {          // (T,B,D) f32
    __shared__ _Float16 As[BM][BK];
    __shared__ _Float16 Bs[BN][BK];
    const int b = blockIdx.z, m0 = blockIdx.y * BM, n0 = blockIdx.x * BN;
    const int tid = threadIdx.x, lane = tid & 31, wave = tid >> 5;
    const int wm = wave >> 1, wn = wave & 1;
    v8f c[2][4];
#pragma unroll
    for (int a = 0; a < 2; ++a)
#pragma unroll
        for (int j = 0; j < 4; ++j) c[a][j] = (v8f){};

    for (int kc = 0; kc < kS; kc += BK) {
#pragma unroll
        for (int i = 0; i < (BM * BK) / 256; ++i) {   // A(m=t,k=s) = Wgt[s][t][b]
            int e = tid + 256 * i, m = e & 127, k = e >> 7;
            As[m][k] = (_Float16)Wgt[(size_t)(kc + k) * (kT * kB) + (size_t)(m0 + m) * kB + b];
        }
#pragma unroll
        for (int i = 0; i < (BN * BK) / 256; ++i) {   // B(k=s,n=d) = SrcH[s][b][d]
            int e = tid + 256 * i, n = e & 127, k = e >> 7;
            Bs[n][k] = SrcH[(size_t)(kc + k) * (kB * kD) + (size_t)b * kD + n0 + n];
        }
        __syncthreads();
        mma_tile(&As[0][0], &Bs[0][0], wm, wn, lane, c);
        __syncthreads();
    }
    const int h = lane >> 4, nl = lane & 15;
#pragma unroll
    for (int ai = 0; ai < 2; ++ai)
#pragma unroll
        for (int j = 0; j < 4; ++j)
#pragma unroll
            for (int r = 0; r < 8; ++r) {
                int t = m0 + wm * 32 + ai * 16 + r + 8 * h;
                int d = n0 + wn * 64 + j * 16 + nl;
                Out[(size_t)t * (kB * kD) + (size_t)b * kD + d] = c[ai][j][r];
            }
}

// ============================================================================
extern "C" void kernel_launch(void* const* d_in, const int* in_sizes, int n_in,
                              void* d_out, int out_size, void* d_ws, size_t ws_size,
                              hipStream_t stream) {
    const float* source = (const float*)d_in[0];   // (S,B,D)
    const float* target = (const float*)d_in[1];   // (T,B,D)
    const float* W1     = (const float*)d_in[2];   // (A,D)
    const float* W2     = (const float*)d_in[3];   // (A,D)
    float* out = (float*)d_out;                    // (T,B,D)

    // workspace: srcH 64MB | tgtH 64MB | W1h 2MB | W2h 2MB | ts 64MB | tt 64MB
    //            | scores f32 128MB   (total 388MB)
    _Float16* srcH = (_Float16*)d_ws;
    _Float16* tgtH = srcH + (size_t)kS * kB * kD;
    _Float16* w1H  = tgtH + (size_t)kT * kB * kD;
    _Float16* w2H  = w1H  + (size_t)kA * kD;
    _Float16* ts   = w2H  + (size_t)kA * kD;
    _Float16* tt   = ts   + (size_t)kB * kS * kA;
    float*    sc   = (float*)(tt + (size_t)kB * kT * kA);

    dim3 blk(256);
    const int nBig = (kS * kB * kD) / 2048, nW = (kA * kD) / 2048;
    cvt_f16<<<dim3(nBig), blk, 0, stream>>>(source, srcH);
    cvt_f16<<<dim3(nBig), blk, 0, stream>>>(target, tgtH);
    cvt_f16<<<dim3(nW),   blk, 0, stream>>>(W1, w1H);
    cvt_f16<<<dim3(nW),   blk, 0, stream>>>(W2, w2H);

    dim3 gGemm(kA / BN, kS / BM, kB);   // (8,8,32) for all 1024x1024 GEMMs
    proj_f16<<<gGemm, blk, 0, stream>>>(srcH, w1H, ts);
    proj_f16<<<gGemm, blk, 0, stream>>>(tgtH, w2H, tt);
    scores_kernel<<<gGemm, blk, 0, stream>>>(ts, tt, sc);
    softmax_s<<<dim3((kT * kB) / 256), blk, 0, stream>>>(sc);
    out_kernel<<<gGemm, blk, 0, stream>>>(sc, srcH, out);
}
